// Net_16793322127774
// MI455X (gfx1250) — matrix operand order repaired and checked by reading, not compiled, not run on hardware
//
#include <hip/hip_runtime.h>
#include <hip/hip_bf16.h>
#include <math.h>
#include <stdint.h>

typedef __attribute__((ext_vector_type(16))) _Float16 v16h;
typedef __attribute__((ext_vector_type(8)))  float    v8f;

#define NB    256          // batch (graphs)
#define NN    256          // nodes per graph (layer 1)
#define FH    128          // F_IN == H == 128
#define NC    6
#define NDEG  16
#define EPG   (NN * NDEG)  // 4096 edges per graph
#define KP1   205
#define KP2   164
#define LS    132          // padded LDS row stride (floats): 528B rows, 16B-aligned
#define M2PAD 208          // K1 padded to 13 WMMA row-tiles
#define WFRAG_MAT (8 * 4 * 32 * 16)   // f16 elems per fragment-swizzled matrix

// ---------- CDNA5 async Global->LDS (ASYNCcnt-tracked DMA) ----------
#if defined(__has_builtin)
#if __has_builtin(__builtin_amdgcn_global_load_async_to_lds_b128)
#define HAVE_ASYNC_LDS 1
#endif
#endif

#if defined(HAVE_ASYNC_LDS)
typedef int v4i_ __attribute__((vector_size(16)));
typedef __attribute__((address_space(1))) v4i_ as1_v4i;
typedef __attribute__((address_space(3))) v4i_ as3_v4i;
#if __has_builtin(__builtin_amdgcn_s_wait_asynccnt)
#define ASYNC_WAIT() __builtin_amdgcn_s_wait_asynccnt(0)
#else
#define ASYNC_WAIT() asm volatile("s_wait_asynccnt 0x0" ::: "memory")
#endif
// copy 16 bytes global -> LDS, no VGPR staging
__device__ __forceinline__ void async_copy16(const float* gp, float* lp) {
  __builtin_amdgcn_global_load_async_to_lds_b128((as1_v4i*)gp, (as3_v4i*)lp, 0, 0);
}
#endif

// ---------------- WMMA fragment helpers ----------------

// A fragment (16x32 f16) built from f32 LDS tile, per documented layout:
// lane<16: row=lane, K = e (e<8) / 16+(e-8) ; lane>=16: row=lane-16, K += 8.
__device__ __forceinline__ v16h a_frag_f32(const float* base, int mi, int kk, int lane) {
  int row  = mi * 16 + (lane & 15);
  int koff = (lane < 16) ? 0 : 8;
  const float* p = base + row * LS + kk * 32 + koff;
  v16h a;
#pragma unroll
  for (int e = 0; e < 8; ++e) a[e] = (_Float16)p[e];
#pragma unroll
  for (int e = 0; e < 8; ++e) a[e + 8] = (_Float16)p[16 + e];
  return a;
}

// 16x16 f32 tile of (ag @ Wrel^T + xs @ Wroot^T), f32 accumulate.
__device__ __forceinline__ v8f gemm_tile(const float* xs, const float* ag,
                                         const _Float16* wf_rel, const _Float16* wf_root,
                                         int mi, int ni, int lane) {
  v8f acc = {};
#pragma unroll
  for (int kk = 0; kk < 4; ++kk) {
    v16h a = a_frag_f32(ag, mi, kk, lane);
    v16h b = *(const v16h*)(wf_rel + (((size_t)ni * 4 + kk) * 32 + lane) * 16);
    acc = __builtin_amdgcn_wmma_f32_16x16x32_f16(false, a, false, b, (short)0, acc, false, false);
  }
#pragma unroll
  for (int kk = 0; kk < 4; ++kk) {
    v16h a = a_frag_f32(xs, mi, kk, lane);
    v16h b = *(const v16h*)(wf_root + (((size_t)ni * 4 + kk) * 32 + lane) * 16);
    acc = __builtin_amdgcn_wmma_f32_16x16x32_f16(false, a, false, b, (short)0, acc, false, false);
  }
  return acc;
}

// Descending bitonic sort of 256 (score, idx) pairs in LDS; all 256 threads call.
__device__ __forceinline__ void bitonic_sort_desc(float* sc, int* sidx, int t) {
  for (int k = 2; k <= 256; k <<= 1) {
    for (int j = k >> 1; j > 0; j >>= 1) {
      int ixj = t ^ j;
      if (ixj > t) {
        bool desc = ((t & k) == 0);
        float va = sc[t], vb = sc[ixj];
        bool sw = desc ? (va < vb) : (va > vb);
        if (sw) {
          sc[t] = vb; sc[ixj] = va;
          int ti = sidx[t]; sidx[t] = sidx[ixj]; sidx[ixj] = ti;
        }
      }
      __syncthreads();
    }
  }
}

// ---------------- prep kernels ----------------

__global__ void pnorm_kernel(const float* __restrict__ p1, const float* __restrict__ p2,
                             float* __restrict__ p1n, float* __restrict__ p2n) {
  __shared__ float red[128];
  const float* p  = (blockIdx.x == 0) ? p1  : p2;
  float*       pn = (blockIdx.x == 0) ? p1n : p2n;
  int t = threadIdx.x;
  float v = p[t];
  red[t] = v * v;
  __syncthreads();
  for (int s = 64; s > 0; s >>= 1) {
    if (t < s) red[t] += red[t + s];
    __syncthreads();
  }
  pn[t] = v * rsqrtf(red[0]);
}

// Swizzle 4 weight matrices [128,128] f32 -> fragment-ordered f16:
// wf16[mat][ni(8)][kk(4)][lane(32)][e(16)], value = W[n][k] with
// n = ni*16 + (lane&15), k = kk*32 + (lane<16?0:8) + (e&7) + (e<8?0:16)   (B = W^T layout).
__global__ void wconv_kernel(const float* __restrict__ W0, const float* __restrict__ W1,
                             const float* __restrict__ W2, const float* __restrict__ W3,
                             _Float16* __restrict__ wf16) {
  int idx = blockIdx.x * 256 + threadIdx.x;      // 65536 total
  int e    = idx & 15;
  int lane = (idx >> 4) & 31;
  int kk   = (idx >> 9) & 3;
  int ni   = (idx >> 11) & 7;
  int m    = (idx >> 14) & 3;
  const float* W = (m == 0) ? W0 : (m == 1) ? W1 : (m == 2) ? W2 : W3;
  int n = ni * 16 + (lane & 15);
  int k = kk * 32 + ((lane < 16) ? 0 : 8) + (e & 7) + ((e < 8) ? 0 : 16);
  wf16[idx] = (_Float16)W[n * FH + k];
}

// ---------------- layer 1: conv1 + relu + topk1 + pool + readout ----------------

__global__ __launch_bounds__(256, 1)
void layer1_kernel(const float* __restrict__ x, const int* __restrict__ src,
                   const int* __restrict__ dst, const float* __restrict__ brel,
                   const _Float16* __restrict__ wf16, const float* __restrict__ p1n,
                   float* __restrict__ hbuf1, float* __restrict__ h1,
                   int* __restrict__ new_idx, float* __restrict__ x1) {
  extern __shared__ float smf[];
  float* xs    = smf;                 // [256][LS]
  float* ag    = xs + NN * LS;        // [256][LS]
  float* pns   = ag + NN * LS;        // [128]
  float* scacc = pns + 128;           // [256]
  float* sc    = scacc + 256;         // [256]
  int*   sidx  = (int*)(sc + 256);    // [256]

  const int g = blockIdx.x, tid = threadIdx.x;
  const int wv = tid >> 5, lane = tid & 31;

  // zero aggregator
  for (int i = tid; i < NN * FH; i += 256) {
    int r = i >> 7, c = i & 127;
    ag[r * LS + c] = 0.f;
  }
  // async DMA this graph's node block into LDS (f32), 16B chunks
#if defined(HAVE_ASYNC_LDS)
  for (int ch = tid; ch < NN * (FH / 4); ch += 256) {
    int r = ch >> 5, c4 = (ch & 31) * 4;
    async_copy16(x + ((size_t)(g * NN + r)) * FH + c4, xs + r * LS + c4);
  }
  ASYNC_WAIT();
#else
  for (int i = tid; i < NN * FH; i += 256) {
    int r = i >> 7, c = i & 127;
    xs[r * LS + c] = x[((size_t)(g * NN + r)) * FH + c];
  }
#endif
  if (tid < 128) pns[tid] = p1n[tid];
  scacc[tid] = 0.f;
  __syncthreads();

  // Edge aggregation fully in LDS: one edge per wave iteration, 4 floats/lane.
  const int* sg = src + g * EPG;
  const int* dg = dst + g * EPG;
  for (int e = wv; e < EPG; e += 8) {
    if (e + 64 < EPG) { __builtin_prefetch(&sg[e + 64], 0, 0); __builtin_prefetch(&dg[e + 64], 0, 0); }
    int s = sg[e] - g * NN;
    int d = dg[e] - g * NN;
    const float* xr = &xs[s * LS];
    float*       ar = &ag[d * LS];
#pragma unroll
    for (int q = 0; q < 4; ++q) {
      int c = lane + q * 32;
      atomicAdd(&ar[c], xr[c]);
    }
  }
  __syncthreads();

  // WMMA GEMM: h = relu(ag@Wrel^T + b + xs@Wroot^T); fused score partials.
  const _Float16* wrel  = wf16 + 0 * WFRAG_MAT;
  const _Float16* wroot = wf16 + 1 * WFRAG_MAT;
  for (int t = wv; t < 16 * 8; t += 8) {
    int mi = t >> 3, ni = t & 7;
    v8f acc = gemm_tile(xs, ag, wrel, wroot, mi, ni, lane);
    int col   = ni * 16 + (lane & 15);
    float bia = brel[col];
    float pw  = pns[col];
    int rbase = mi * 16 + ((lane < 16) ? 0 : 8);
#pragma unroll
    for (int r = 0; r < 8; ++r) {
      float v = fmaxf(acc[r] + bia, 0.f);
      int row = rbase + r;
      hbuf1[((size_t)(g * NN + row)) * FH + col] = v;
      atomicAdd(&scacc[row], v * pw);      // score = h . (p/||p||)
    }
  }
  __syncthreads();

  // topk: score -> sort 256 descending.
  sc[tid] = tanhf(scacc[tid]);
  sidx[tid] = tid;
  __syncthreads();
  bitonic_sort_desc(sc, sidx, tid);

  // new node ids (kept: rank; dropped: -1)
  {
    int node = sidx[tid];
    new_idx[g * NN + node] = (tid < KP1) ? (g * KP1 + tid) : -1;
  }
  __syncthreads();

  // x_new = h[kept] * score; column-wise max/mean readout.
  int c = tid & 127, half = tid >> 7;
  float mx = -INFINITY, sm = 0.f;
  for (int r = half; r < KP1; r += 2) {
    int node = sidx[r];
    float v = hbuf1[((size_t)(g * NN + node)) * FH + c] * sc[r];
    h1[((size_t)(g * KP1 + r)) * FH + c] = v;
    mx = fmaxf(mx, v);
    sm += v;
  }
  scacc[tid] = mx;
  __syncthreads();
  if (tid < 128) x1[g * 256 + tid] = fmaxf(scacc[tid], scacc[tid + 128]);
  __syncthreads();
  scacc[tid] = sm;
  __syncthreads();
  if (tid < 128) x1[g * 256 + 128 + tid] = (scacc[tid] + scacc[tid + 128]) * (1.0f / KP1);
}

// ---------------- layer 2: conv2 (remapped edges) + relu + topk2 + readout ----------------

__global__ __launch_bounds__(256, 1)
void layer2_kernel(const int* __restrict__ src, const int* __restrict__ dst,
                   const int* __restrict__ new_idx, const float* __restrict__ brel,
                   const _Float16* __restrict__ wf16, const float* __restrict__ p2n,
                   const float* __restrict__ h1, float* __restrict__ hbuf2,
                   float* __restrict__ x2) {
  extern __shared__ float smf[];
  float* xs    = smf;                  // [208][LS] (rows >= K1 zero)
  float* ag    = xs + M2PAD * LS;      // [208][LS]
  float* pns   = ag + M2PAD * LS;      // [128]
  float* scacc = pns + 128;            // [256]
  float* sc    = scacc + 256;          // [256]
  int*   sidx  = (int*)(sc + 256);     // [256]

  const int g = blockIdx.x, tid = threadIdx.x;
  const int wv = tid >> 5, lane = tid & 31;

  // zero aggregator + pad rows of xs
  for (int i = tid; i < M2PAD * FH; i += 256) {
    int r = i >> 7, c = i & 127;
    ag[r * LS + c] = 0.f;
    if (r >= KP1) xs[r * LS + c] = 0.f;
  }
  // async DMA pooled features into LDS
#if defined(HAVE_ASYNC_LDS)
  for (int ch = tid; ch < KP1 * (FH / 4); ch += 256) {
    int r = ch >> 5, c4 = (ch & 31) * 4;
    async_copy16(h1 + ((size_t)(g * KP1 + r)) * FH + c4, xs + r * LS + c4);
  }
  ASYNC_WAIT();
#else
  for (int i = tid; i < KP1 * FH; i += 256) {
    int r = i >> 7, c = i & 127;
    xs[r * LS + c] = h1[((size_t)(g * KP1 + r)) * FH + c];
  }
#endif
  if (tid < 128) pns[tid] = p2n[tid];
  scacc[tid] = 0.f;
  __syncthreads();

  // Edge remap + masked aggregation (edges of graph g are e in [g*EPG, (g+1)*EPG)).
  const int* sg = src + g * EPG;
  const int* dg = dst + g * EPG;
  for (int e = wv; e < EPG; e += 8) {
    int s2 = new_idx[sg[e]];
    int d2 = new_idx[dg[e]];
    if ((s2 | d2) >= 0) {              // both kept
      int sl = s2 - g * KP1, dl = d2 - g * KP1;
      const float* xr = &xs[sl * LS];
      float*       ar = &ag[dl * LS];
#pragma unroll
      for (int q = 0; q < 4; ++q) {
        int c = lane + q * 32;
        atomicAdd(&ar[c], xr[c]);
      }
    }
  }
  __syncthreads();

  const _Float16* wrel  = wf16 + 2 * WFRAG_MAT;
  const _Float16* wroot = wf16 + 3 * WFRAG_MAT;
  for (int t = wv; t < 13 * 8; t += 8) {
    int mi = t >> 3, ni = t & 7;
    v8f acc = gemm_tile(xs, ag, wrel, wroot, mi, ni, lane);
    int col   = ni * 16 + (lane & 15);
    float bia = brel[col];
    float pw  = pns[col];
    int rbase = mi * 16 + ((lane < 16) ? 0 : 8);
#pragma unroll
    for (int r = 0; r < 8; ++r) {
      int row = rbase + r;
      if (row < KP1) {
        float v = fmaxf(acc[r] + bia, 0.f);
        hbuf2[((size_t)(g * KP1 + row)) * FH + col] = v;
        atomicAdd(&scacc[row], v * pw);
      }
    }
  }
  __syncthreads();

  sc[tid]   = (tid < KP1) ? tanhf(scacc[tid]) : -INFINITY;  // pad sinks to bottom
  sidx[tid] = tid;
  __syncthreads();
  bitonic_sort_desc(sc, sidx, tid);

  // readout over top-K2 scaled rows (pooled features only feed the readout).
  int c = tid & 127, half = tid >> 7;
  float mx = -INFINITY, sm = 0.f;
  for (int r = half; r < KP2; r += 2) {
    int node = sidx[r];
    float v = hbuf2[((size_t)(g * KP1 + node)) * FH + c] * sc[r];
    mx = fmaxf(mx, v);
    sm += v;
  }
  scacc[tid] = mx;
  __syncthreads();
  if (tid < 128) x2[g * 256 + tid] = fmaxf(scacc[tid], scacc[tid + 128]);
  __syncthreads();
  scacc[tid] = sm;
  __syncthreads();
  if (tid < 128) x2[g * 256 + 128 + tid] = (scacc[tid] + scacc[tid + 128]) * (1.0f / KP2);
}

// ---------------- MLP head: z=x1+x2 -> lin1+relu -> lin2 -> log_softmax ----------------

__global__ __launch_bounds__(128, 1)
void mlp_kernel(const float* __restrict__ x1, const float* __restrict__ x2,
                const float* __restrict__ Wl1, const float* __restrict__ bl1,
                const float* __restrict__ Wl2, const float* __restrict__ bl2,
                float* __restrict__ out) {
  __shared__ float z[256];
  __shared__ float z1[128];
  __shared__ float lg[NC];
  int g = blockIdx.x, t = threadIdx.x;
  z[t]       = x1[g * 256 + t]       + x2[g * 256 + t];
  z[t + 128] = x1[g * 256 + 128 + t] + x2[g * 256 + 128 + t];
  __syncthreads();
  float acc = bl1[t];
  for (int k = 0; k < 256; ++k) acc += Wl1[t * 256 + k] * z[k];
  z1[t] = fmaxf(acc, 0.f);
  __syncthreads();
  if (t < NC) {
    float a = bl2[t];
    for (int k = 0; k < 128; ++k) a += Wl2[t * 128 + k] * z1[k];
    lg[t] = a;
  }
  __syncthreads();
  if (t == 0) {
    float m = lg[0];
    for (int cix = 1; cix < NC; ++cix) m = fmaxf(m, lg[cix]);
    float s = 0.f;
    for (int cix = 0; cix < NC; ++cix) s += expf(lg[cix] - m);
    float L = logf(s);
    for (int cix = 0; cix < NC; ++cix) out[g * NC + cix] = lg[cix] - m - L;
  }
}

// ---------------- host launcher ----------------

extern "C" void kernel_launch(void* const* d_in, const int* in_sizes, int n_in,
                              void* d_out, int out_size, void* d_ws, size_t ws_size,
                              hipStream_t stream) {
  const float* x      = (const float*)d_in[0];
  const int*   src    = (const int*)d_in[1];
  const int*   dst    = (const int*)d_in[2];
  const float* W1rel  = (const float*)d_in[3];
  const float* b1rel  = (const float*)d_in[4];
  const float* W1root = (const float*)d_in[5];
  const float* p1     = (const float*)d_in[6];
  const float* W2rel  = (const float*)d_in[7];
  const float* b2rel  = (const float*)d_in[8];
  const float* W2root = (const float*)d_in[9];
  const float* p2     = (const float*)d_in[10];
  const float* Wl1    = (const float*)d_in[11];
  const float* bl1    = (const float*)d_in[12];
  const float* Wl2    = (const float*)d_in[13];
  const float* bl2    = (const float*)d_in[14];
  float* out = (float*)d_out;

  // workspace carve (wf16 first for 32B-aligned v16h loads)
  char* w = (char*)d_ws;
  size_t off = 0;
  _Float16* wf16 = (_Float16*)(w + off); off += (size_t)4 * WFRAG_MAT * sizeof(_Float16);
  float* p1n     = (float*)(w + off);    off += 128 * sizeof(float);
  float* p2n     = (float*)(w + off);    off += 128 * sizeof(float);
  int*   new_idx = (int*)(w + off);      off += (size_t)NB * NN * sizeof(int);
  float* x1      = (float*)(w + off);    off += (size_t)NB * 256 * sizeof(float);
  float* x2      = (float*)(w + off);    off += (size_t)NB * 256 * sizeof(float);
  float* hbuf1   = (float*)(w + off);    off += (size_t)NB * NN * FH * sizeof(float);
  float* h1      = (float*)(w + off);    off += (size_t)NB * KP1 * FH * sizeof(float);
  float* hbuf2   = (float*)(w + off);    off += (size_t)NB * KP1 * FH * sizeof(float);
  (void)ws_size; (void)n_in; (void)in_sizes; (void)out_size;

  const size_t sm1 = ((size_t)(2 * NN * LS) + 128 + 256 + 256) * sizeof(float) + 256 * sizeof(int);
  const size_t sm2 = ((size_t)(2 * M2PAD * LS) + 128 + 256 + 256) * sizeof(float) + 256 * sizeof(int);

  pnorm_kernel<<<2, 128, 0, stream>>>(p1, p2, p1n, p2n);
  wconv_kernel<<<256, 256, 0, stream>>>(W1rel, W1root, W2rel, W2root, wf16);
  layer1_kernel<<<NB, 256, sm1, stream>>>(x, src, dst, b1rel, wf16, p1n,
                                          hbuf1, h1, new_idx, x1);
  layer2_kernel<<<NB, 256, sm2, stream>>>(src, dst, new_idx, b2rel, wf16, p2n,
                                          h1, hbuf2, x2);
  mlp_kernel<<<NB, 128, 0, stream>>>(x1, x2, Wl1, bl1, Wl2, bl2, out);
}
